// MultiHeadAttention_83880711291583
// MI455X (gfx1250) — compile-verified, matrix operand order checked by reading, not run on hardware
//
#include <hip/hip_runtime.h>
#include <hip/hip_bf16.h>
#include <math.h>

// ---- problem constants (match reference) ----
static constexpr int B_ = 4, S_ = 2048, D_ = 768, H_ = 12, E_ = 64;
static constexpr float LN_EPS_ = 1e-5f;

typedef __bf16 bf16_t;
typedef __attribute__((ext_vector_type(16))) __bf16 v16bf;
typedef __attribute__((ext_vector_type(8)))  __bf16 v8bf;   // 16-byte pack
typedef __attribute__((ext_vector_type(8)))  float  v8f;

// -------------------------------------------------------------------------
// Fragment loader for V_WMMA_F32_16X16X32_BF16 (wave32):
//   A 16x32 (MxK): lane holds row (lane&15), 16 contiguous K at k0+(lane>>4)*16
//   B 32x16 (KxN): lane holds col (lane&15) with K contiguous -> load from the
//                  TRANSPOSED matrix stored row-major with leading dim `ld`.
// -------------------------------------------------------------------------
__device__ __forceinline__ v16bf load_frag(const bf16_t* base, int ld, int r0,
                                           int k0, int lane) {
  const int m  = lane & 15;
  const int kk = k0 + ((lane >> 4) << 4);
  return *(const v16bf*)(base + (size_t)(r0 + m) * ld + kk);
}

__device__ __forceinline__ v8f wmma_bf16(v16bf a, v16bf b, v8f c) {
  return __builtin_amdgcn_wmma_f32_16x16x32_bf16(false, a, false, b,
                                                 (short)0, c, false, false);
}

// ---------------- conversion / transpose kernels ----------------
__global__ __launch_bounds__(256) void k_cvt_bf16(const float* __restrict__ x,
                                                  bf16_t* __restrict__ y, int n) {
  for (int i = blockIdx.x * blockDim.x + threadIdx.x; i < n;
       i += gridDim.x * blockDim.x)
    y[i] = (bf16_t)x[i];
}

// W [H,D,E] -> WT [H,E,D]
__global__ __launch_bounds__(256) void k_tr_hde(const float* __restrict__ W,
                                                bf16_t* __restrict__ WT) {
  int n = H_ * D_ * E_;
  for (int i = blockIdx.x * blockDim.x + threadIdx.x; i < n;
       i += gridDim.x * blockDim.x) {
    int h = i / (D_ * E_);
    int r = i - h * (D_ * E_);
    int d = r / E_, e = r - d * E_;
    WT[((size_t)h * E_ + e) * D_ + d] = (bf16_t)W[i];
  }
}

// Wo [768,768] -> WoT [768,768]
__global__ __launch_bounds__(256) void k_tr_wo(const float* __restrict__ W,
                                               bf16_t* __restrict__ WT) {
  int n = D_ * D_;
  for (int i = blockIdx.x * blockDim.x + threadIdx.x; i < n;
       i += gridDim.x * blockDim.x) {
    int f = i / D_, d = i - f * D_;
    WT[(size_t)d * D_ + f] = (bf16_t)W[i];
  }
}

// ---------------- QKV projection: one wave per (b,h,s-tile16) ----------------
// q,k stored row-major bf16 [B,H,S,E] (staged via LDS -> b128 stores);
// v stored TRANSPOSED bf16 [B,H,E,S] (contiguous b128 stores from registers).
__global__ __launch_bounds__(32) void
k_qkv(const bf16_t* __restrict__ xb, const bf16_t* __restrict__ wqT,
      const bf16_t* __restrict__ wkT, const bf16_t* __restrict__ wvT,
      const float* __restrict__ bq, const float* __restrict__ bk,
      const float* __restrict__ bv, bf16_t* __restrict__ q,
      bf16_t* __restrict__ k, bf16_t* __restrict__ vT) {
  __shared__ bf16_t st[16 * E_];            // 2 KB staging tile

  const int lane = threadIdx.x & 31;
  const int tile = blockIdx.x;
  const int ts   = S_ / 16;                 // 128
  const int s0   = (tile % ts) << 4;
  const int h    = (tile / ts) % H_;
  const int bz   = tile / (ts * H_);

  const bf16_t* xrow = xb + (size_t)bz * S_ * D_;
  const bf16_t* wq   = wqT + (size_t)h * E_ * D_;
  const bf16_t* wk   = wkT + (size_t)h * E_ * D_;
  const bf16_t* wv   = wvT + (size_t)h * E_ * D_;

  v8f aq[4], ak[4], av[4];
#pragma unroll
  for (int i = 0; i < 4; ++i) { aq[i] = v8f{}; ak[i] = v8f{}; av[i] = v8f{}; }

  for (int kc = 0; kc < D_ / 32; ++kc) {          // 24 K-chunks
    const int k0 = kc * 32;
    __builtin_prefetch(xrow + (size_t)(s0 + (lane & 15)) * D_ + k0 + 64, 0, 1);
    v16bf a = load_frag(xrow, D_, s0, k0, lane);
#pragma unroll
    for (int nt = 0; nt < 4; ++nt) {
      const int n0 = nt * 16;
      aq[nt] = wmma_bf16(a, load_frag(wq, D_, n0, k0, lane), aq[nt]);
      ak[nt] = wmma_bf16(a, load_frag(wk, D_, n0, k0, lane), ak[nt]);
      av[nt] = wmma_bf16(a, load_frag(wv, D_, n0, k0, lane), av[nt]);
    }
  }

  const int N     = lane & 15;
  const int Mbase = (lane >> 4) << 3;
  const size_t bh = (size_t)bz * H_ + h;

  // ---- v: pack 8 rows per lane -> one b128 store (vT is [B,H,E,S]) ----
#pragma unroll
  for (int nt = 0; nt < 4; ++nt) {
    const float bi = bv[h * E_ + nt * 16 + N];
    v8bf pk;
#pragma unroll
    for (int r = 0; r < 8; ++r) pk[r] = (bf16_t)(av[nt][r] + bi);
    *(v8bf*)(vT + (bh * E_ + nt * 16 + N) * S_ + s0 + Mbase) = pk;
  }

  // ---- q then k: stage 16x64 tile in LDS, flush as contiguous b128 ----
#pragma unroll
  for (int mat = 0; mat < 2; ++mat) {
    const v8f*   acc  = mat ? ak : aq;
    const float* bias = mat ? bk : bq;
    bf16_t*      dst  = mat ? k : q;
#pragma unroll
    for (int nt = 0; nt < 4; ++nt) {
      const float bi = bias[h * E_ + nt * 16 + N];
#pragma unroll
      for (int r = 0; r < 8; ++r)
        st[(Mbase + r) * E_ + nt * 16 + N] = (bf16_t)(acc[nt][r] + bi);
    }
    __syncthreads();
#pragma unroll
    for (int c = 0; c < 4; ++c) {                 // 32 lanes * 4 * 8 elems
      const int idx = (c * 32 + lane) * 8;
      const int row = idx >> 6, col = idx & 63;
      *(v8bf*)(dst + (bh * S_ + s0 + row) * E_ + col) =
          *(const v8bf*)(st + idx);
    }
    __syncthreads();
  }
}

// ---------------- attention: one wave per (b,h,s-tile16) ----------------
// Phase 1: scaled scores (2 WMMAs per 16x16 t-tile, K=E=64) -> f32 LDS
// Phase 2: coalesced mask apply + exact softmax per row -> bf16 P in LDS
// Phase 3: ctx = P x V via WMMA; A-frags from LDS (ds), B-frags from vT;
//          output tile staged in LDS and flushed as b128 stores.
__global__ __launch_bounds__(32) void
k_attn(const bf16_t* __restrict__ q, const bf16_t* __restrict__ k,
       const bf16_t* __restrict__ vT, const int* __restrict__ mask,
       bf16_t* __restrict__ ctx) {
  extern __shared__ __align__(32) char smem[];        // 16*2048 f32 = 128 KB
  float* sS = (float*)smem;
  __shared__ bf16_t sP[16 * S_];                      // 64 KB static LDS

  const int lane = threadIdx.x & 31;
  const int tile = blockIdx.x;
  const int ts   = S_ / 16;
  const int s0   = (tile % ts) << 4;
  const int h    = (tile / ts) % H_;
  const int bz   = tile / (ts * H_);
  const size_t bh = (size_t)bz * H_ + h;

  const bf16_t* qbh = q + bh * S_ * E_;
  const bf16_t* kbh = k + bh * S_ * E_;
  const bf16_t* vbh = vT + bh * E_ * S_;

  const v16bf qa0 = load_frag(qbh, E_, s0, 0, lane);
  const v16bf qa1 = load_frag(qbh, E_, s0, 32, lane);
  const float scale = rsqrtf((float)S_);              // ref scales by 1/sqrt(S)

  const int N     = lane & 15;
  const int Mbase = (lane >> 4) << 3;

  // ---- phase 1: scaled scores into LDS (mask deferred to phase 2) ----
  for (int tt = 0; tt < ts; ++tt) {
    const int t0 = tt * 16;
    __builtin_prefetch(kbh + (size_t)(t0 + 16 + (lane & 15)) * E_, 0, 1);
    v8f acc = v8f{};
    acc = wmma_bf16(qa0, load_frag(kbh, E_, t0, 0, lane), acc);
    acc = wmma_bf16(qa1, load_frag(kbh, E_, t0, 32, lane), acc);
#pragma unroll
    for (int r = 0; r < 8; ++r)
      sS[(Mbase + r) * S_ + t0 + N] = acc[r] * scale;
  }
  __syncthreads();

  // ---- phase 2: coalesced mask + exact softmax per row ----
  for (int row = 0; row < 16; ++row) {
    const int* mrow = mask + ((size_t)bz * S_ + s0 + row) * S_;
    float mx = -3.4e38f;
    for (int c = lane; c < S_; c += 32) {             // mask: coalesced b32
      float v = sS[row * S_ + c];
      if (mrow[c]) v = -1e9f;
      sS[row * S_ + c] = v;
      mx = fmaxf(mx, v);
    }
#pragma unroll
    for (int off = 16; off > 0; off >>= 1) mx = fmaxf(mx, __shfl_xor(mx, off, 32));
    float sum = 0.f;
    for (int c = lane; c < S_; c += 32) {
      float e = __expf(sS[row * S_ + c] - mx);
      sS[row * S_ + c] = e;                           // cache exp for pass 3
      sum += e;
    }
#pragma unroll
    for (int off = 16; off > 0; off >>= 1) sum += __shfl_xor(sum, off, 32);
    const float inv = 1.f / sum;
    for (int c = lane; c < S_; c += 32)
      sP[row * S_ + c] = (bf16_t)(sS[row * S_ + c] * inv);
  }
  __syncthreads();

  // ---- phase 3: ctx = P (LDS) x V (vT, contiguous B-frags) ----
  v8f o[4];
#pragma unroll
  for (int i = 0; i < 4; ++i) o[i] = v8f{};
  for (int kc = 0; kc < S_ / 32; ++kc) {              // 64 K-chunks
    const int k0 = kc * 32;
    v16bf a = load_frag(sP, S_, 0, k0, lane);         // ds_load path
#pragma unroll
    for (int nt = 0; nt < 4; ++nt)
      o[nt] = wmma_bf16(a, load_frag(vbh, S_, nt * 16, k0, lane), o[nt]);
  }
  __syncthreads();

  // stage ctx tile (16x64) in LDS, flush as contiguous b128 stores
  bf16_t* sc = sP;                                    // reuse front of sP
#pragma unroll
  for (int nt = 0; nt < 4; ++nt)
#pragma unroll
    for (int r = 0; r < 8; ++r)
      sc[(Mbase + r) * E_ + nt * 16 + N] = (bf16_t)o[nt][r];
  __syncthreads();
#pragma unroll
  for (int c = 0; c < 4; ++c) {
    const int idx = (c * 32 + lane) * 8;
    const int row = idx >> 6, col = idx & 63;
    *(v8bf*)(ctx + ((size_t)bz * S_ + s0 + row) * (H_ * E_) + h * E_ + col) =
        *(const v8bf*)(sc + idx);
  }
}

// ---------------- output projection + LayerNorm ----------------
// 128 threads (4 waves); block owns 16 rows of [B*S, 768]. Each wave does 12
// n-tiles; result tile staged in LDS for row-wise LayerNorm.
__global__ __launch_bounds__(128) void
k_proj_ln(const bf16_t* __restrict__ ctx, const bf16_t* __restrict__ woT,
          const float* __restrict__ bo, const float* __restrict__ gamma,
          const float* __restrict__ beta, float* __restrict__ out) {
  __shared__ float sOut[16 * D_];                     // 48 KB
  __shared__ float sMu[16], sRs[16];

  const int tid  = threadIdx.x;
  const int lane = tid & 31;
  const int wave = tid >> 5;
  const int row0 = blockIdx.x * 16;                   // row in [0, B*S)
  const bf16_t* arow = ctx + (size_t)row0 * D_;

  v8f acc[12];
#pragma unroll
  for (int i = 0; i < 12; ++i) acc[i] = v8f{};

  for (int kc = 0; kc < D_ / 32; ++kc) {              // 24 K-chunks
    const int k0 = kc * 32;
    v16bf a = load_frag(arow, D_, 0, k0, lane);
#pragma unroll
    for (int j = 0; j < 12; ++j) {
      const int n0 = (wave * 12 + j) * 16;
      acc[j] = wmma_bf16(a, load_frag(woT, D_, n0, k0, lane), acc[j]);
    }
  }

  const int N     = lane & 15;
  const int Mbase = (lane >> 4) << 3;
#pragma unroll
  for (int j = 0; j < 12; ++j) {
    const int n0 = (wave * 12 + j) * 16;
    const float bi = bo[n0 + N];
#pragma unroll
    for (int r = 0; r < 8; ++r)
      sOut[(Mbase + r) * D_ + n0 + N] = acc[j][r] + bi;
  }
  __syncthreads();

  if (tid < 16) {
    float mu = 0.f;
    for (int c = 0; c < D_; ++c) mu += sOut[tid * D_ + c];
    mu *= (1.f / D_);
    float var = 0.f;
    for (int c = 0; c < D_; ++c) {
      float d = sOut[tid * D_ + c] - mu;
      var += d * d;
    }
    var *= (1.f / D_);
    sMu[tid] = mu;
    sRs[tid] = rsqrtf(var + LN_EPS_);
  }
  __syncthreads();

  for (int idx = tid; idx < 16 * D_; idx += 128) {
    const int r = idx / D_, c = idx - r * D_;
    out[(size_t)row0 * D_ + idx] =
        (sOut[idx] - sMu[r]) * sRs[r] * gamma[c] + beta[c];
  }
}

// ---------------- host launcher ----------------
extern "C" void kernel_launch(void* const* d_in, const int* in_sizes, int n_in,
                              void* d_out, int out_size, void* d_ws, size_t ws_size,
                              hipStream_t stream) {
  const float* x    = (const float*)d_in[0];
  const int*   mask = (const int*)d_in[1];
  const float* Wq   = (const float*)d_in[2];
  const float* bq   = (const float*)d_in[3];
  const float* Wk   = (const float*)d_in[4];
  const float* bk   = (const float*)d_in[5];
  const float* Wv   = (const float*)d_in[6];
  const float* bv   = (const float*)d_in[7];
  const float* Wo   = (const float*)d_in[8];
  const float* bo   = (const float*)d_in[9];
  const float* gamma = (const float*)d_in[10];
  const float* beta  = (const float*)d_in[11];
  float* out = (float*)d_out;

  // workspace layout (bf16 intermediates), ~64.5 MB total
  char* ws = (char*)d_ws;
  const size_t szX  = (size_t)B_ * S_ * D_ * sizeof(bf16_t);
  const size_t szW  = (size_t)H_ * E_ * D_ * sizeof(bf16_t);
  const size_t szWo = (size_t)D_ * D_ * sizeof(bf16_t);
  const size_t szQ  = (size_t)B_ * H_ * S_ * E_ * sizeof(bf16_t);
  bf16_t* xb  = (bf16_t*)(ws);
  bf16_t* wqT = (bf16_t*)(ws + szX);
  bf16_t* wkT = (bf16_t*)(ws + szX + szW);
  bf16_t* wvT = (bf16_t*)(ws + szX + 2 * szW);
  bf16_t* woT = (bf16_t*)(ws + szX + 3 * szW);
  bf16_t* q   = (bf16_t*)(ws + szX + 3 * szW + szWo);
  bf16_t* k   = (bf16_t*)(ws + szX + 3 * szW + szWo + szQ);
  bf16_t* vT  = (bf16_t*)(ws + szX + 3 * szW + szWo + 2 * szQ);
  bf16_t* ctx = (bf16_t*)(ws + szX + 3 * szW + szWo + 3 * szQ);

  const int nX = B_ * S_ * D_;
  k_cvt_bf16<<<(nX + 255) / 256, 256, 0, stream>>>(x, xb, nX);
  k_tr_hde<<<(H_ * D_ * E_ + 255) / 256, 256, 0, stream>>>(Wq, wqT);
  k_tr_hde<<<(H_ * D_ * E_ + 255) / 256, 256, 0, stream>>>(Wk, wkT);
  k_tr_hde<<<(H_ * D_ * E_ + 255) / 256, 256, 0, stream>>>(Wv, wvT);
  k_tr_wo<<<(D_ * D_ + 255) / 256, 256, 0, stream>>>(Wo, woT);

  const int nTiles = B_ * H_ * (S_ / 16);             // 6144
  k_qkv<<<nTiles, 32, 0, stream>>>(xb, wqT, wkT, wvT, bq, bk, bv, q, k, vT);

  const size_t attnShmem = (size_t)16 * S_ * sizeof(float);   // 128 KB dynamic
  k_attn<<<nTiles, 32, attnShmem, stream>>>(q, k, vT, mask, ctx);

  k_proj_ln<<<B_ * S_ / 16, 128, 0, stream>>>(ctx, woT, bo, gamma, beta, out);
}
